// DIMKT_47785806135762
// MI455X (gfx1250) — compile-verified, matrix operand order checked by reading, not run on hardware
//
#include <hip/hip_runtime.h>
#include <hip/hip_bf16.h>
#include <math.h>

// ---------------- CDNA5 WMMA types ----------------
typedef __attribute__((ext_vector_type(16))) _Float16 v16h;
typedef __attribute__((ext_vector_type(8)))  _Float16 v8h;
typedef __attribute__((ext_vector_type(8)))  float    v8f;

#define E_DIM 256
#define B_DIM 128
#define T_DIM 200
#define L_DIM 201
#define NROWS 25600           // B*T
#define NBLK  16              // persistent workgroups in recurrence

// ---------------- fragment helpers (ISA 7.12.2 layouts, wave32) ----------------
// A (16x32 f16): lane L holds row M=L%16; halves h=0..7 -> K=(L>=16?8:0)+h,
// h=8..15 -> K=16+(L>=16?8:0)+(h-8).  Two contiguous 8-half runs per lane.
__device__ __forceinline__ v16h frag_a_f16(const _Float16* rowp, int k0, int lane) {
  const _Float16* p = rowp + k0 + ((lane >> 4) << 3);
  v8h lo = *(const v8h*)(p);
  v8h hi = *(const v8h*)(p + 16);
  v16h a;
#pragma unroll
  for (int i = 0; i < 8; ++i) { a[i] = lo[i]; a[i + 8] = hi[i]; }
  return a;
}

// Same A fragment but gathered from an f32 row (q_table only; single use per row).
__device__ __forceinline__ v16h frag_a_f32(const float* rowp, int k0, int lane) {
  const float* p = rowp + k0 + ((lane >> 4) << 3);
  v16h a;
#pragma unroll
  for (int i = 0; i < 8; ++i) { a[i] = (_Float16)p[i]; a[i + 8] = (_Float16)p[i + 16]; }
  return a;
}

// B (32x16 f16) fragments PRE-PACKED per (nt,kt) tile: 32 lanes x 16 halves
// contiguous -> one 32-byte load per lane (global flavor, used in big GEMM).
__device__ __forceinline__ v16h frag_b(const _Float16* wp, int ktiles, int nt, int kt, int lane) {
  const _Float16* p = wp + ((((size_t)nt * ktiles + kt) * 32 + lane) << 4);
  return *(const v16h*)p;
}

// LDS flavor: slice base already selected for this WG's nt.
__device__ __forceinline__ v16h frag_b_lds(const _Float16* slice, int kt, int lane) {
  const _Float16* p = slice + (((kt << 5) + lane) << 4);
  v8h lo = *(const v8h*)(p);
  v8h hi = *(const v8h*)(p + 8);
  v16h b;
#pragma unroll
  for (int i = 0; i < 8; ++i) { b[i] = lo[i]; b[i + 8] = hi[i]; }
  return b;
}

__device__ __forceinline__ v8f wmma_f16(v16h a, v16h b, v8f c) {
  return __builtin_amdgcn_wmma_f32_16x16x32_f16(false, a, false, b, (short)0, c, false, false);
}

__device__ __forceinline__ float sigmf(float x) { return 1.f / (1.f + __expf(-x)); }

// ---------------- inter-WG ticket barrier (monotonic counter) ----------------
__device__ __forceinline__ void gbar(int* bar) {
  __threadfence();
  __syncthreads();
  if (threadIdx.x == 0) {
    int ticket = __hip_atomic_fetch_add(bar, 1, __ATOMIC_ACQ_REL, __HIP_MEMORY_SCOPE_AGENT);
    int target = (ticket / NBLK + 1) * NBLK;
    while (__hip_atomic_load(bar, __ATOMIC_ACQUIRE, __HIP_MEMORY_SCOPE_AGENT) < target)
      __builtin_amdgcn_s_sleep(2);
  }
  __syncthreads();
  __threadfence();
}

// ---------------- kernel 0: pack W (K,N f32 row-major) into B-fragment layout f16 ----------------
__global__ void dimkt_pack_w(const float* __restrict__ W, _Float16* __restrict__ out, int K, int N) {
  int idx = blockIdx.x * blockDim.x + threadIdx.x;
  if (idx >= K * N) return;
  int h    = idx & 15;
  int lane = (idx >> 4) & 31;
  int tile = idx >> 9;                // nt * Ktiles + kt
  int ktiles = K >> 5;
  int kt = tile % ktiles, nt = tile / ktiles;
  int k = kt * 32 + (h & 7) + ((h >> 3) << 4) + ((lane >> 4) << 3);
  int n = nt * 16 + (lane & 15);
  out[idx] = (_Float16)W[(size_t)k * N + n];
}

// ---------------- kernel 0b: f32 -> f16 table conversion ----------------
__global__ void dimkt_cvt(const float* __restrict__ src, _Float16* __restrict__ dst, int n) {
  int i = blockIdx.x * blockDim.x + threadIdx.x;
  if (i < n) dst[i] = (_Float16)src[i];
}

// ---------------- kernel 1: init k0 = broadcast(knowledge), reset barrier ----------------
__global__ void dimkt_init(const float* __restrict__ knowledge, _Float16* __restrict__ kbuf,
                           int* __restrict__ bar) {
  int i = blockIdx.x * blockDim.x + threadIdx.x;   // 0..32767
  kbuf[i] = (_Float16)knowledge[i & (E_DIM - 1)];
  if (i == 0) *bar = 0;
}

// ---------------- kernel 2: input_data / target_data = gather-concat @ W1 + b1 ----------------
// grid.x = 3200 M-tiles over 51200 logical rows (first 25600 -> input, rest -> target, shifted idx)
// blockDim = 512 (16 waves), wave w owns N-tile w.  K loop split per source table: no branches.
__global__ void dimkt_embed_gemm(
    const int* __restrict__ questions, const int* __restrict__ skills,
    const int* __restrict__ qdiff, const int* __restrict__ sdiff,
    const float* __restrict__ q_table, const _Float16* __restrict__ c16,
    const _Float16* __restrict__ sd16, const _Float16* __restrict__ qd16,
    const _Float16* __restrict__ w1p, const float* __restrict__ b1,
    _Float16* __restrict__ in_f16, _Float16* __restrict__ tg_f16)
{
  const int lane  = threadIdx.x & 31;
  const int nt    = threadIdx.x >> 5;           // 0..15
  const int mtile = blockIdx.x;                 // 0..3199
  const int rowg  = mtile * 16 + (lane & 15);   // 0..51199
  const bool tgt  = rowg >= NROWS;
  const int r     = tgt ? rowg - NROWS : rowg;
  const int b     = r / T_DIM;
  const int t     = r % T_DIM + (tgt ? 1 : 0);
  const int base  = b * L_DIM + t;

  const float*    q_row  = q_table + (size_t)questions[base] * E_DIM;
  const _Float16* c_row  = c16  + (size_t)skills[base] * E_DIM;
  const _Float16* sd_row = sd16 + (size_t)sdiff[base]  * E_DIM;
  const _Float16* qd_row = qd16 + (size_t)qdiff[base]  * E_DIM;

  v8f acc = {};
#pragma unroll 4
  for (int kt = 0; kt < 8; ++kt)
    acc = wmma_f16(frag_a_f32(q_row, kt * 32, lane), frag_b(w1p, 32, nt, kt, lane), acc);
#pragma unroll 4
  for (int kt = 0; kt < 8; ++kt)
    acc = wmma_f16(frag_a_f16(c_row, kt * 32, lane), frag_b(w1p, 32, nt, 8 + kt, lane), acc);
#pragma unroll 4
  for (int kt = 0; kt < 8; ++kt)
    acc = wmma_f16(frag_a_f16(sd_row, kt * 32, lane), frag_b(w1p, 32, nt, 16 + kt, lane), acc);
#pragma unroll 4
  for (int kt = 0; kt < 8; ++kt)
    acc = wmma_f16(frag_a_f16(qd_row, kt * 32, lane), frag_b(w1p, 32, nt, 24 + kt, lane), acc);

  const int n    = nt * 16 + (lane & 15);
  const float bi = b1[n];
  _Float16* dst  = tgt ? tg_f16 : in_f16;
  const int mro  = mtile * 16 + ((lane >> 4) << 3);   // D rows: mro+v
#pragma unroll
  for (int v = 0; v < 8; ++v) {
    int m  = mro + v;
    int mr = tgt ? m - NROWS : m;
    dst[(size_t)mr * E_DIM + n] = (_Float16)(acc[v] + bi);
  }
}

// ---------------- kernel 3: the sequential scan (persistent, 16 WGs x 8 waves) ----------------
// WG nt owns output cols [nt*16, nt*16+16); wave mt owns rows [mt*16, mt*16+16).
// W2..W6 slices (80KB f16) resident in LDS; k/a fragments cached in VGPRs per step;
// all K-loops are straight-line (no per-iteration branching).
__global__ void __launch_bounds__(256)
dimkt_recur(
    const int* __restrict__ responses, const int* __restrict__ sdiff, const int* __restrict__ qdiff,
    const _Float16* __restrict__ a16, const _Float16* __restrict__ sd16, const _Float16* __restrict__ qd16,
    const _Float16* __restrict__ w2p, const _Float16* __restrict__ w3p,
    const _Float16* __restrict__ w4p, const _Float16* __restrict__ w5p,
    const _Float16* __restrict__ w6p,
    const float* __restrict__ b2, const float* __restrict__ b3, const float* __restrict__ b4,
    const float* __restrict__ b5, const float* __restrict__ b6,
    const _Float16* __restrict__ in_f16,
    _Float16* __restrict__ kbuf,     // ping-pong: 2 * (128*256)
    _Float16* __restrict__ sdfbuf,   // 128*256
    _Float16* __restrict__ out_f16,  // (B*T, E)
    int* __restrict__ bar)
{
  const int lane  = threadIdx.x & 31;
  const int mt    = threadIdx.x >> 5;            // 0..7
  const int nt    = blockIdx.x;                  // 0..15
  const int arow  = mt * 16 + (lane & 15);       // batch row for A fragments
  const int n     = nt * 16 + (lane & 15);       // output col
  const int mro   = mt * 16 + ((lane >> 4) << 3);// D row base
  const float bb2 = b2[n], bb3 = b3[n], bb4 = b4[n], bb5 = b5[n], bb6 = b6[n];

  // ---- stage this WG's weight slices into LDS (packed layout preserved) ----
  __shared__ _Float16 wlds[40960];               // 80 KB
  _Float16* w2s = wlds;                          //  8 tiles * 512
  _Float16* w3s = wlds + 4096;
  _Float16* w4s = wlds + 8192;                   // 16 tiles * 512
  _Float16* w5s = wlds + 16384;
  _Float16* w6s = wlds + 24576;                  // 32 tiles * 512
  {
    const _Float16* gsrc[5] = { w2p + (size_t)nt * 4096,  w3p + (size_t)nt * 4096,
                                w4p + (size_t)nt * 8192,  w5p + (size_t)nt * 8192,
                                w6p + (size_t)nt * 16384 };
    _Float16*       ldst[5] = { w2s, w3s, w4s, w5s, w6s };
    const int       lens[5] = { 4096, 4096, 8192, 8192, 16384 };
#pragma unroll
    for (int s = 0; s < 5; ++s)
      for (int i = threadIdx.x * 8; i < lens[s]; i += 256 * 8)
        *(v8h*)(ldst[s] + i) = *(const v8h*)(gsrc[s] + i);
  }
  __syncthreads();

  for (int t = 0; t < T_DIM; ++t) {
    const _Float16* kcur = kbuf + (size_t)(t & 1) * (B_DIM * E_DIM);
    _Float16*       knxt = kbuf + (size_t)((t + 1) & 1) * (B_DIM * E_DIM);
    const int ibase = arow * L_DIM + t;
    int rr = responses[ibase]; rr = (rr > 0) ? 1 : 0;     // masked response
    const _Float16* a_row  = a16  + (size_t)rr * E_DIM;
    const _Float16* sd_row = sd16 + (size_t)sdiff[ibase] * E_DIM;
    const _Float16* qd_row = qd16 + (size_t)qdiff[ibase] * E_DIM;
    const _Float16* krow = kcur + (size_t)arow * E_DIM;
    const _Float16* irow = in_f16 + ((size_t)arow * T_DIM + t) * E_DIM;

    // ---- phase A: SDFt = sigmoid(qq@W2+b2) * tanh(qq@W3+b3), qq = k - input ----
    // k fragments cached for reuse in the ins-GEMM (K 0..255).
    v16h kfr[8];
    v8f acc2 = {}, acc3 = {};
#pragma unroll
    for (int kt = 0; kt < 8; ++kt) {
      kfr[kt] = frag_a_f16(krow, kt * 32, lane);
      v16h qq = kfr[kt] - frag_a_f16(irow, kt * 32, lane);
      acc2 = wmma_f16(qq, frag_b_lds(w2s, kt, lane), acc2);
      acc3 = wmma_f16(qq, frag_b_lds(w3s, kt, lane), acc3);
    }
#pragma unroll
    for (int v = 0; v < 8; ++v) {
      float s = sigmf(acc2[v] + bb2) * tanhf(acc3[v] + bb3);
      sdfbuf[(size_t)(mro + v) * E_DIM + n] = (_Float16)s;
    }
    gbar(bar);

    // ---- phase B: PKA from x2=[SDF,a];  KSU from ins=[k,a,sd,qd];  k update ----
    // a_t fragments cached: used by both the x2-GEMM and the ins-GEMM.
    v16h afr[8];
#pragma unroll
    for (int kt = 0; kt < 8; ++kt) afr[kt] = frag_a_f16(a_row, kt * 32, lane);

    v8f acc4 = {}, acc5 = {}, acc6 = {};
    const _Float16* srow = sdfbuf + (size_t)arow * E_DIM;
#pragma unroll
    for (int kt = 0; kt < 8; ++kt) {
      v16h a = frag_a_f16(srow, kt * 32, lane);
      acc4 = wmma_f16(a, frag_b_lds(w4s, kt, lane), acc4);
      acc5 = wmma_f16(a, frag_b_lds(w5s, kt, lane), acc5);
    }
#pragma unroll
    for (int kt = 0; kt < 8; ++kt) {
      acc4 = wmma_f16(afr[kt], frag_b_lds(w4s, 8 + kt, lane), acc4);
      acc5 = wmma_f16(afr[kt], frag_b_lds(w5s, 8 + kt, lane), acc5);
    }
#pragma unroll
    for (int kt = 0; kt < 8; ++kt)
      acc6 = wmma_f16(kfr[kt], frag_b_lds(w6s, kt, lane), acc6);
#pragma unroll
    for (int kt = 0; kt < 8; ++kt)
      acc6 = wmma_f16(afr[kt], frag_b_lds(w6s, 8 + kt, lane), acc6);
#pragma unroll
    for (int kt = 0; kt < 8; ++kt)
      acc6 = wmma_f16(frag_a_f16(sd_row, kt * 32, lane), frag_b_lds(w6s, 16 + kt, lane), acc6);
#pragma unroll
    for (int kt = 0; kt < 8; ++kt)
      acc6 = wmma_f16(frag_a_f16(qd_row, kt * 32, lane), frag_b_lds(w6s, 24 + kt, lane), acc6);

#pragma unroll
    for (int v = 0; v < 8; ++v) {
      const int m = mro + v;
      float pk   = sigmf(acc4[v] + bb4) * tanhf(acc5[v] + bb5);
      float gk   = sigmf(acc6[v] + bb6);
      float kold = (float)kcur[(size_t)m * E_DIM + n];
      float kn   = gk * kold + (1.f - gk) * pk;
      knxt[(size_t)m * E_DIM + n] = (_Float16)kn;
      out_f16[((size_t)m * T_DIM + t) * E_DIM + n] = (_Float16)kn;
    }
    gbar(bar);
  }
}

// ---------------- kernel 4: pred = sigmoid(sum_e target*output) ----------------
__global__ void dimkt_pred(const _Float16* __restrict__ tg, const _Float16* __restrict__ ot,
                           float* __restrict__ pred) {
  const int lane = threadIdx.x & 31;
  const int wv   = threadIdx.x >> 5;
  const int idx  = blockIdx.x * 8 + wv;            // flat (b*T + t)
  const _Float16* tp = tg + (size_t)idx * E_DIM + lane * 8;
  const _Float16* op = ot + (size_t)idx * E_DIM + lane * 8;
  float s = 0.f;
#pragma unroll
  for (int i = 0; i < 8; ++i) s += (float)tp[i] * (float)op[i];
#pragma unroll
  for (int off = 16; off > 0; off >>= 1) s += __shfl_xor(s, off, 32);
  if (lane == 0) pred[idx] = sigmf(s);
}

// ---------------- host side ----------------
extern "C" void kernel_launch(void* const* d_in, const int* in_sizes, int n_in,
                              void* d_out, int out_size, void* d_ws, size_t ws_size,
                              hipStream_t stream) {
  const int*   questions = (const int*)  d_in[0];
  const int*   skills    = (const int*)  d_in[1];
  const int*   qdiff     = (const int*)  d_in[2];   // question_difficulty
  const int*   sdiff     = (const int*)  d_in[3];   // skill_difficulty
  const int*   responses = (const int*)  d_in[4];
  const float* q_table   = (const float*)d_in[5];
  const float* c_table   = (const float*)d_in[6];
  const float* sd_table  = (const float*)d_in[7];
  const float* qd_table  = (const float*)d_in[8];
  const float* a_table   = (const float*)d_in[9];
  const float* knowledge = (const float*)d_in[10];
  const float* W1 = (const float*)d_in[11]; const float* b1 = (const float*)d_in[12];
  const float* W2 = (const float*)d_in[13]; const float* b2 = (const float*)d_in[14];
  const float* W3 = (const float*)d_in[15]; const float* b3 = (const float*)d_in[16];
  const float* W4 = (const float*)d_in[17]; const float* b4 = (const float*)d_in[18];
  const float* W5 = (const float*)d_in[19]; const float* b5 = (const float*)d_in[20];
  const float* W6 = (const float*)d_in[21]; const float* b6 = (const float*)d_in[22];

  // workspace layout (offsets in _Float16 elements)
  _Float16* ws  = (_Float16*)d_ws;
  _Float16* w1p = ws;                         // 1024x256 packed
  _Float16* w2p = ws + 262144;                // 256x256
  _Float16* w3p = ws + 327680;                // 256x256
  _Float16* w4p = ws + 393216;                // 512x256
  _Float16* w5p = ws + 524288;                // 512x256
  _Float16* w6p = ws + 655360;                // 1024x256
  _Float16* in_f16  = ws + 917504;            // 25600x256
  _Float16* tg_f16  = ws + 7471104;           // 25600x256
  _Float16* out_f16 = ws + 14024704;          // 25600x256
  _Float16* kbuf    = ws + 20578304;          // 2 x 128x256
  _Float16* sdfbuf  = ws + 20643840;          // 128x256
  _Float16* c16     = ws + 20676608;          // 201x256
  _Float16* sd16    = ws + 20728064;          // 102x256
  _Float16* qd16    = ws + 20754176;          // 102x256
  _Float16* a16     = ws + 20780288;          // 2x256
  int* bar = (int*)((char*)d_ws + 41561600);  // past all f16 buffers

  // phase 0: pack weights into WMMA B-fragment layout (f16); convert hot tables
  dimkt_pack_w<<<(1024 * 256 + 255) / 256, 256, 0, stream>>>(W1, w1p, 1024, 256);
  dimkt_pack_w<<<( 256 * 256 + 255) / 256, 256, 0, stream>>>(W2, w2p,  256, 256);
  dimkt_pack_w<<<( 256 * 256 + 255) / 256, 256, 0, stream>>>(W3, w3p,  256, 256);
  dimkt_pack_w<<<( 512 * 256 + 255) / 256, 256, 0, stream>>>(W4, w4p,  512, 256);
  dimkt_pack_w<<<( 512 * 256 + 255) / 256, 256, 0, stream>>>(W5, w5p,  512, 256);
  dimkt_pack_w<<<(1024 * 256 + 255) / 256, 256, 0, stream>>>(W6, w6p, 1024, 256);
  dimkt_cvt<<<(201 * 256 + 255) / 256, 256, 0, stream>>>(c_table,  c16,  201 * 256);
  dimkt_cvt<<<(102 * 256 + 255) / 256, 256, 0, stream>>>(sd_table, sd16, 102 * 256);
  dimkt_cvt<<<(102 * 256 + 255) / 256, 256, 0, stream>>>(qd_table, qd16, 102 * 256);
  dimkt_cvt<<<(  2 * 256 + 255) / 256, 256, 0, stream>>>(a_table,  a16,    2 * 256);

  // phase 1: init state + barrier
  dimkt_init<<<128, 256, 0, stream>>>(knowledge, kbuf, bar);

  // phase 2: input_data / target_data GEMMs (gathered A, WMMA)
  dimkt_embed_gemm<<<3200, 512, 0, stream>>>(questions, skills, qdiff, sdiff,
                                             q_table, c16, sd16, qd16,
                                             w1p, b1, in_f16, tg_f16);

  // phase 3: sequential scan, 16 persistent WGs, weights resident in LDS
  dimkt_recur<<<NBLK, 256, 0, stream>>>(responses, sdiff, qdiff,
                                        a16, sd16, qd16,
                                        w2p, w3p, w4p, w5p, w6p,
                                        b2, b3, b4, b5, b6,
                                        in_f16, kbuf, sdfbuf, out_f16, bar);

  // phase 4: predictions
  dimkt_pred<<<3200, 256, 0, stream>>>(tg_f16, out_f16, (float*)d_out);
}
